// Cross_Attention_18425409700231
// MI455X (gfx1250) — matrix operand order repaired and checked by reading, not compiled, not run on hardware
//
#include <hip/hip_runtime.h>
#include <stdint.h>

// ---------------- types ----------------
typedef __attribute__((ext_vector_type(16))) __bf16 v16bf;
typedef __attribute__((ext_vector_type(8)))  float  v8f;
typedef __attribute__((ext_vector_type(2)))  float  v2f;

union Frag16 { v16bf v; uint4 q[2]; };

__device__ __forceinline__ uint16_t f2bf(float f) {
  uint32_t u = __float_as_uint(f);
  uint32_t r = u + 0x7FFFu + ((u >> 16) & 1u);   // round-to-nearest-even
  return (uint16_t)(r >> 16);
}

#ifndef __has_builtin
#define __has_builtin(x) 0
#endif
#if __has_builtin(__builtin_amdgcn_tensor_load_to_lds) && __has_builtin(__builtin_amdgcn_s_wait_tensorcnt)
#define USE_TDM 1
#else
#define USE_TDM 0
#endif

#if USE_TDM
typedef __attribute__((ext_vector_type(4))) unsigned tdm_g0_t;
typedef __attribute__((ext_vector_type(8))) int      tdm_g1_t;
typedef __attribute__((ext_vector_type(4))) int      tdm_g2_t;

// TDM: copy n_elem bf16 (contiguous in global) into LDS at lds_byte_off,
// inserting a 16B skip after every 256B so the LDS layout has a padded
// stride of 136 uint16 per 128-channel pixel (bank-conflict padding done
// by the DMA engine, not by VALU/DS traffic).
//   pad_interval code 5 -> every 64 dwords; pad_amount code 3 -> 4 dwords.
__device__ __forceinline__ void tdm_load_row_padded(uint32_t lds_byte_off,
                                                    const void* gptr,
                                                    uint32_t n_elem) {
  uint64_t ga = (uint64_t)(uintptr_t)gptr;
  tdm_g0_t g0;
  g0.x = 1u;                                      // count=1 (user descriptor)
  g0.y = lds_byte_off;                            // lds_addr
  g0.z = (uint32_t)ga;                            // global_addr[31:0]
  g0.w = (uint32_t)(ga >> 32) | (2u << 30);       // global_addr[56:32] | type=2
  tdm_g1_t g1;
  g1[0] = (int)((1u << 16) | (1u << 20) | (5u << 22) | (3u << 25));
  //        data_size=2B   pad_enable    interval=64dw  amount=4dw ; wg_mask=0
  g1[1] = (int)(n_elem << 16);                    // tensor_dim0[15:0] @ bits 63:48
  g1[2] = (int)((n_elem >> 16) | (1u << 16));     // tensor_dim0 hi | tensor_dim1=1
  g1[3] = (int)(n_elem << 16);                    // tile_dim0 @ bits 127:112
  g1[4] = 1;                                      // tile_dim1=1, tile_dim2=0
  g1[5] = (int)n_elem;                            // tensor_dim0_stride lo32
  g1[6] = 0;
  g1[7] = 0;
  tdm_g2_t gz = {0, 0, 0, 0};
#if __clang_major__ >= 23
  tdm_g1_t gz8 = {0, 0, 0, 0, 0, 0, 0, 0};
  __builtin_amdgcn_tensor_load_to_lds(g0, g1, gz, gz, gz8, 0);
#else
  __builtin_amdgcn_tensor_load_to_lds(g0, g1, gz, gz, 0);
#endif
}
#endif // USE_TDM

// ---------------- problem constants ----------------
#define BATCH 8
#define CCH   128
#define NPIX  16384

// ---------------- workspace layout (bytes) ----------------
#define OFF_RNORM 0u
#define OFF_ATTN  4096u
#define OFF_AO    36864u
#define OFF_Y1    (OFF_AO  + 33554432u)
#define OFF_Y2    (OFF_Y1  + 33554432u)
#define OFF_SC    (OFF_Y2  + 67108864u)
#define OFF_W1P   (OFF_SC  + 67108864u)
#define OFF_W2P   (OFF_W1P + 294912u)
#define OFF_SCP   (OFF_W2P + 32768u)

// =====================================================================
// Weight packing into B fragments for v_wmma_f32_16x16x32_bf16, in this code's own contiguous K order (A loads match).
// Lane l holds K = (l>>4)*16 + j (j=0..15), N = l&15.
// =====================================================================
__global__ void pack3x3_kernel(const float* __restrict__ w, uint16_t* __restrict__ wp) {
  int L = blockIdx.x * 256 + threadIdx.x;          // 147456
  if (L >= 147456) return;
  int j = L & 15, lane = (L >> 4) & 31, nt = (L >> 9) & 7, kk = L >> 12;
  int k  = kk * 32 + ((lane >> 4) * 16) + j;
  int ky = k / 384, kx = (k % 384) / 128, ci = k & 127;
  int co = nt * 16 + (lane & 15);
  wp[L] = f2bf(w[((co * 128 + ci) * 3 + ky) * 3 + kx]);
}

__global__ void pack1x1_kernel(const float* __restrict__ w, uint16_t* __restrict__ wp) {
  int L = blockIdx.x * 256 + threadIdx.x;          // 16384
  if (L >= 16384) return;
  int j = L & 15, lane = (L >> 4) & 31, nt = (L >> 9) & 7, kk = L >> 12;
  int ci = kk * 32 + ((lane >> 4) * 16) + j;
  int co = nt * 16 + (lane & 15);
  wp[L] = f2bf(w[co * 128 + ci]);
}

// =====================================================================
// 1) Row L2 norms
// =====================================================================
__global__ __launch_bounds__(256) void rownorm_kernel(const float* __restrict__ x1,
                                                      float* __restrict__ rnorm) {
  int row = blockIdx.x;
  const float* p = x1 + (size_t)row * NPIX;
  float s = 0.f;
  for (int i = threadIdx.x; i < NPIX; i += 256) { float v = p[i]; s += v * v; }
  __shared__ float red[256];
  red[threadIdx.x] = s; __syncthreads();
  for (int o = 128; o > 0; o >>= 1) {
    if (threadIdx.x < (unsigned)o) red[threadIdx.x] += red[threadIdx.x + o];
    __syncthreads();
  }
  if (threadIdx.x == 0) rnorm[row] = 1.f / fmaxf(sqrtf(red[0]), 1e-12f);
}

// =====================================================================
// 2) Gram (v_wmma_f32_16x16x4_f32) + temperature + softmax
// =====================================================================
__global__ __launch_bounds__(256) void gram_softmax_kernel(const float* __restrict__ x1,
                                                           const float* __restrict__ rnorm,
                                                           const float* __restrict__ temperature,
                                                           float* __restrict__ attn) {
  int bh = blockIdx.x;
  int b = bh >> 3, hp = bh & 7;
  int lane = threadIdx.x & 31, wave = threadIdx.x >> 5;
  int m = lane & 15, hi = lane >> 4;
  const float* qrow = x1 + ((size_t)(b * 128 + hp * 16 + m)) * NPIX;

  v8f c = {0.f, 0.f, 0.f, 0.f, 0.f, 0.f, 0.f, 0.f};
  int k0 = wave * 2048;
  for (int k = k0; k < k0 + 2048; k += 4) {
    v2f a;
    a.x = qrow[k + hi * 2 + 0];
    a.y = qrow[k + hi * 2 + 1];
    c = __builtin_amdgcn_wmma_f32_16x16x4_f32(false, a, false, a, (short)0, c, false, false);
  }

  __shared__ float part[8][16][16];
  #pragma unroll
  for (int r = 0; r < 8; r++) part[wave][r + hi * 8][m] = c[r];
  __syncthreads();

  if (threadIdx.x < 16) {
    int row = threadIdx.x;
    int head = hp * 2 + (row >> 3);
    float rm = rnorm[b * 128 + hp * 16 + row];
    float temp = temperature[head];
    int nb = (row >> 3) * 8;
    float vals[8]; float mx = -1e30f;
    #pragma unroll
    for (int j = 0; j < 8; j++) {
      int n = nb + j;
      float s = 0.f;
      #pragma unroll
      for (int w = 0; w < 8; w++) s += part[w][row][n];
      float v = s * rm * rnorm[b * 128 + hp * 16 + n] * temp;
      vals[j] = v; mx = fmaxf(mx, v);
    }
    float den = 0.f;
    #pragma unroll
    for (int j = 0; j < 8; j++) { vals[j] = __expf(vals[j] - mx); den += vals[j]; }
    float inv = 1.f / den;
    float* ap = attn + ((size_t)(b * 16 + head)) * 64 + (row & 7) * 8;
    #pragma unroll
    for (int j = 0; j < 8; j++) ap[j] = vals[j] * inv;
  }
}

// =====================================================================
// 3) attn x V -> ao (NHWC bf16), LDS transpose
// =====================================================================
__global__ __launch_bounds__(256) void apply_attn_kernel(const float* __restrict__ x2,
                                                         const float* __restrict__ attn,
                                                         uint16_t* __restrict__ ao) {
  int b = blockIdx.y, n0 = blockIdx.x * 32;
  __shared__ float attnS[1024];
  __shared__ float vt[128][33];
  for (int i = threadIdx.x; i < 1024; i += 256) attnS[i] = attn[b * 1024 + i];
  const float* vb = x2 + (size_t)b * CCH * NPIX + n0;
  for (int i = threadIdx.x; i < 128 * 32; i += 256) {
    int ch = i >> 5, nn = i & 31;
    vt[ch][nn] = vb[(size_t)ch * NPIX + nn];
  }
  __syncthreads();
  uint16_t* aob = ao + (size_t)b * NPIX * CCH + (size_t)n0 * CCH;
  for (int i = threadIdx.x; i < 128 * 32; i += 256) {
    int ch = i & 127, nn = i >> 7;
    int h = ch >> 3;
    const float* arow = attnS + h * 64 + (ch & 7) * 8;
    float s = 0.f;
    #pragma unroll
    for (int d = 0; d < 8; d++) s += arow[d] * vt[h * 8 + d][nn];
    aob[(size_t)nn * CCH + ch] = f2bf(s);
  }
}

// =====================================================================
// 4) 3x3 conv (zero pad) implicit GEMM, bf16 WMMA, TDM-staged LDS rows,
//    fused bias+LeakyReLU. OOB ky rows are skipped (zero padding).
// =====================================================================
__global__ __launch_bounds__(256) void conv1_kernel(const uint16_t* __restrict__ ao,
                                                    const uint16_t* __restrict__ w1p,
                                                    const float* __restrict__ bias,
                                                    uint16_t* __restrict__ y1) {
  int by = blockIdx.x; int b = by >> 7; int y = by & 127;
  __shared__ uint16_t aoS[130 * 136];              // cols x=-1..128, stride 136
  int lane = threadIdx.x & 31, w = threadIdx.x >> 5;
  int m = lane & 15, hi = lane >> 4;
  int x0 = w * 16;

  // prefetch packed weights into cache hierarchy (global_prefetch_b8)
  __builtin_prefetch(w1p + (size_t)threadIdx.x * 576, 0, 0);

  // zero the two edge columns once (TDM pad-skip never writes them)
  for (int i = threadIdx.x; i < 34; i += 256) {
    int col = (i < 17) ? 0 : 129;
    int cs = (i < 17) ? i : (i - 17);
    uint4 z = {0u, 0u, 0u, 0u};
    *(uint4*)(&aoS[col * 136 + cs * 8]) = z;
  }

  const v8f vz = {0.f, 0.f, 0.f, 0.f, 0.f, 0.f, 0.f, 0.f};
  v8f acc[8];
  #pragma unroll
  for (int nt = 0; nt < 8; nt++) acc[nt] = vz;

#if USE_TDM
  uint32_t lds_base = (uint32_t)(uintptr_t)(&aoS[0]);
#endif

  for (int ky = 0; ky < 3; ky++) {
    int yy = y + ky - 1;
    if ((unsigned)yy >= 128u) continue;            // zero-padded row: contributes 0
    __syncthreads();                               // prior LDS reads / edge-zero done
#if USE_TDM
    if (threadIdx.x < 32) {
      tdm_load_row_padded(lds_base + 272u,
                          ao + ((size_t)(b * 128 + yy)) * 128 * 128,
                          128 * 128);
      __builtin_amdgcn_s_wait_tensorcnt(0);
    }
#else
    for (int i = threadIdx.x; i < 128 * 16; i += 256) {
      int px = i >> 4, cs = i & 15;
      *(uint4*)(&aoS[(px + 1) * 136 + cs * 8]) =
          *(const uint4*)(ao + ((size_t)((b * 128 + yy) * 128 + px) * 128 + cs * 8));
    }
#endif
    __syncthreads();
    for (int kc = 0; kc < 12; kc++) {
      int kx = kc >> 2, cb = (kc & 3) * 32;
      int kk = ky * 12 + kc;
      Frag16 a;
      const uint16_t* ab = &aoS[(x0 + m + kx) * 136 + cb];
      a.q[0] = *(const uint4*)(ab + hi * 16);
      a.q[1] = *(const uint4*)(ab + hi * 16 + 8);
      const uint16_t* wb = w1p + ((size_t)kk * 4096 + (size_t)lane * 16);
      #pragma unroll
      for (int nt = 0; nt < 8; nt++) {
        Frag16 bf;
        bf.q[0] = *(const uint4*)(wb + nt * 512);
        bf.q[1] = *(const uint4*)(wb + nt * 512 + 8);
        acc[nt] = __builtin_amdgcn_wmma_f32_16x16x32_bf16(
            false, a.v, false, bf.v, (short)0, acc[nt], false, false);
      }
    }
  }

  uint16_t* orow = y1 + (size_t)(b * 128 + y) * 128 * 128;
  #pragma unroll
  for (int nt = 0; nt < 8; nt++) {
    int co = nt * 16 + m;
    float bs = bias[co];
    #pragma unroll
    for (int r = 0; r < 8; r++) {
      int x = x0 + r + hi * 8;                     // D: M = r + hi*8, N = lane&15
      float v = acc[nt][r] + bs;
      v = v >= 0.f ? v : 0.2f * v;
      orow[(size_t)x * 128 + co] = f2bf(v);
    }
  }
}

// =====================================================================
// 5) 1x1 conv GEMM (conv2 + shortcut), bf16 WMMA, TDM-staged input row
// =====================================================================
__global__ __launch_bounds__(256) void gemm1x1_kernel(const uint16_t* __restrict__ in,
                                                      const uint16_t* __restrict__ wp,
                                                      const float* __restrict__ bias,
                                                      float* __restrict__ out) {
  int by = blockIdx.x;
  __shared__ uint16_t s[128 * 136];
  const uint16_t* src = in + (size_t)by * 128 * 128;

  __builtin_prefetch(wp + (size_t)threadIdx.x * 64, 0, 0);

#if USE_TDM
  if (threadIdx.x < 32) {
    tdm_load_row_padded((uint32_t)(uintptr_t)(&s[0]), src, 128 * 128);
    __builtin_amdgcn_s_wait_tensorcnt(0);
  }
#else
  for (int i = threadIdx.x; i < 128 * 16; i += 256) {
    int px = i >> 4, cs = i & 15;
    *(uint4*)(&s[px * 136 + cs * 8]) = *(const uint4*)(src + px * 128 + cs * 8);
  }
#endif
  __syncthreads();

  int lane = threadIdx.x & 31, w = threadIdx.x >> 5;
  int m = lane & 15, hi = lane >> 4, x0 = w * 16;

  const v8f vz = {0.f, 0.f, 0.f, 0.f, 0.f, 0.f, 0.f, 0.f};
  v8f acc[8];
  #pragma unroll
  for (int nt = 0; nt < 8; nt++) acc[nt] = vz;

  #pragma unroll
  for (int kk = 0; kk < 4; kk++) {
    Frag16 a;
    const uint16_t* ab = &s[(x0 + m) * 136 + kk * 32];
    a.q[0] = *(const uint4*)(ab + hi * 16);
    a.q[1] = *(const uint4*)(ab + hi * 16 + 8);
    const uint16_t* wb = wp + (size_t)kk * 4096 + (size_t)lane * 16;
    #pragma unroll
    for (int nt = 0; nt < 8; nt++) {
      Frag16 bf;
      bf.q[0] = *(const uint4*)(wb + nt * 512);
      bf.q[1] = *(const uint4*)(wb + nt * 512 + 8);
      acc[nt] = __builtin_amdgcn_wmma_f32_16x16x32_bf16(
          false, a.v, false, bf.v, (short)0, acc[nt], false, false);
    }
  }
  float* ob = out + (size_t)by * 128 * 128;
  #pragma unroll
  for (int nt = 0; nt < 8; nt++) {
    int co = nt * 16 + m;
    float bs = bias[co];
    #pragma unroll
    for (int r = 0; r < 8; r++) {
      int x = x0 + r + hi * 8;
      ob[(size_t)x * 128 + co] = acc[nt][r] + bs;
    }
  }
}

// =====================================================================
// 6) depthwise 3x3 (reflect) + bias + LeakyReLU + shortcut + residual
// =====================================================================
__global__ __launch_bounds__(256) void dw_final_kernel(const float* __restrict__ y2,
                                                       const float* __restrict__ sc,
                                                       const float* __restrict__ x1,
                                                       const float* __restrict__ dww,
                                                       const float* __restrict__ dwb,
                                                       float* __restrict__ out) {
  int x0 = blockIdx.x * 16;
  int y = blockIdx.y;
  int b = blockIdx.z;
  __shared__ float y2S[3 * 128 * 18];
  __shared__ float scS[16 * 130];

  for (int i = threadIdx.x; i < 3 * 18 * 128; i += 256) {
    int dy = i / (18 * 128); int rm = i % (18 * 128);
    int col = rm >> 7; int co = rm & 127;
    int yy = y + dy - 1; yy = yy < 0 ? 1 : (yy > 127 ? 126 : yy);
    int xx = x0 + col - 1; xx = xx < 0 ? 1 : (xx > 127 ? 126 : xx);
    y2S[(dy * 128 + co) * 18 + col] =
        y2[((size_t)((b * 128 + yy) * 128 + xx)) * 128 + co];
  }
  for (int i = threadIdx.x; i < 16 * 128; i += 256) {
    int x = i >> 7, co = i & 127;
    scS[x * 130 + co] = sc[((size_t)((b * 128 + y) * 128 + x0 + x)) * 128 + co];
  }
  __syncthreads();

  for (int i = threadIdx.x; i < 16 * 128; i += 256) {
    int x = i & 15, co = i >> 4;
    float a = dwb[co];
    #pragma unroll
    for (int dy = 0; dy < 3; dy++)
      #pragma unroll
      for (int dx = 0; dx < 3; dx++)
        a += dww[co * 9 + dy * 3 + dx] * y2S[(dy * 128 + co) * 18 + x + dx];
    a = a >= 0.f ? a : 0.2f * a;
    size_t oi = ((size_t)(b * 128 + co) * 128 + y) * 128 + (x0 + x);
    out[oi] = a + scS[x * 130 + co] + x1[oi];
  }
}

// =====================================================================
extern "C" void kernel_launch(void* const* d_in, const int* in_sizes, int n_in,
                              void* d_out, int out_size, void* d_ws, size_t ws_size,
                              hipStream_t stream) {
  (void)in_sizes; (void)n_in; (void)out_size; (void)ws_size;
  const float* x1          = (const float*)d_in[0];
  const float* x2          = (const float*)d_in[1];
  const float* temperature = (const float*)d_in[2];
  const float* conv1_w     = (const float*)d_in[3];
  const float* conv1_b     = (const float*)d_in[4];
  const float* conv2_w     = (const float*)d_in[5];
  const float* conv2_b     = (const float*)d_in[6];
  const float* dw_w        = (const float*)d_in[7];
  const float* dw_b        = (const float*)d_in[8];
  const float* sc_w        = (const float*)d_in[9];
  const float* sc_b        = (const float*)d_in[10];

  char* ws = (char*)d_ws;
  float*    rnorm = (float*)(ws + OFF_RNORM);
  float*    attn  = (float*)(ws + OFF_ATTN);
  uint16_t* ao    = (uint16_t*)(ws + OFF_AO);
  uint16_t* y1    = (uint16_t*)(ws + OFF_Y1);
  float*    y2    = (float*)(ws + OFF_Y2);
  float*    sc    = (float*)(ws + OFF_SC);
  uint16_t* w1p   = (uint16_t*)(ws + OFF_W1P);
  uint16_t* w2p   = (uint16_t*)(ws + OFF_W2P);
  uint16_t* scp   = (uint16_t*)(ws + OFF_SCP);
  float*    out   = (float*)d_out;

  pack3x3_kernel<<<576, 256, 0, stream>>>(conv1_w, w1p);
  pack1x1_kernel<<<64, 256, 0, stream>>>(conv2_w, w2p);
  pack1x1_kernel<<<64, 256, 0, stream>>>(sc_w, scp);
  rownorm_kernel<<<1024, 256, 0, stream>>>(x1, rnorm);
  gram_softmax_kernel<<<64, 256, 0, stream>>>(x1, rnorm, temperature, attn);
  apply_attn_kernel<<<dim3(512, 8), 256, 0, stream>>>(x2, attn, ao);
  conv1_kernel<<<1024, 256, 0, stream>>>(ao, w1p, conv1_b, y1);
  gemm1x1_kernel<<<1024, 256, 0, stream>>>(y1, w2p, conv2_b, y2);
  gemm1x1_kernel<<<1024, 256, 0, stream>>>(ao, scp, sc_b, sc);
  dw_final_kernel<<<dim3(8, 128, 8), 256, 0, stream>>>(y2, sc, x1, dw_w, dw_b, out);
}
